// ComirecSA_4303557231217
// MI455X (gfx1250) — matrix operand order repaired and checked by reading, not compiled, not run on hardware
//
#include <hip/hip_runtime.h>

typedef __attribute__((ext_vector_type(16))) _Float16 v16h;
typedef __attribute__((ext_vector_type(8)))  _Float16 v8h;
typedef __attribute__((ext_vector_type(8)))  float    v8f;

static __device__ __forceinline__ v8f wmma16(v16h a, v16h b, v8f c) {
  // D = A(16x32 f16) * B(32x16 f16) + C(16x16 f32)
  return __builtin_amdgcn_wmma_f32_16x16x32_f16(false, a, false, b, (short)0, c,
                                                false, false);
}

// Branchless tanh: v_med3 clamp + v_exp_f32 + v_rcp_f32 + 3 VALU.
// ~1ulp reciprocal is far below the f16 quantization of the WMMA inputs.
static __device__ __forceinline__ float fast_tanh(float x) {
  float xc = fminf(fmaxf(x, -15.f), 15.f);
  float e = __expf(2.f * xc);
  return (e - 1.f) * __builtin_amdgcn_rcpf(e + 1.f);
}

// A-matrix fragment (16x32 f16, row-major source in LDS, row stride rs halves).
// Lane l: row = m0 + (l&15); 16 contiguous K halves starting at kb + (l>>4)*16.
static __device__ __forceinline__ v16h load_a(const _Float16* base, int m0, int kb, int rs) {
  int l = threadIdx.x & 31;
  const _Float16* p = base + (m0 + (l & 15)) * rs + kb + ((l >> 4) << 4);
  union { v16h v; v8h h[2]; } u;
  u.h[0] = *(const v8h*)(p);
  u.h[1] = *(const v8h*)(p + 8);
  return u.v;
}

// B-matrix fragment (32x16 f16) from column-major storage (BT[n][k], row stride rs).
// Lane l: n = n0 + (l&15); 16 contiguous K halves starting at kb + (l>>4)*16.
static __device__ __forceinline__ v16h load_b(const _Float16* base, int n0, int kb, int rs) {
  int l = threadIdx.x & 31;
  const _Float16* p = base + (n0 + (l & 15)) * rs + kb + ((l >> 4) << 4);
  union { v16h v; v8h h[2]; } u;
  u.h[0] = *(const v8h*)(p);
  u.h[1] = *(const v8h*)(p + 8);
  return u.v;
}

// B-matrix fragment gathered from a ROW-major [K][N] LDS tile (strided b16 reads).
static __device__ __forceinline__ v16h load_b_rowmajor(const _Float16* base, int n0, int kb, int rs) {
  int l = threadIdx.x & 31;
  int n = n0 + (l & 15);
  int k0 = kb + ((l >> 4) << 4);
  union { v16h v; _Float16 h[16]; } u;
#pragma unroll
  for (int j = 0; j < 16; ++j) u.h[j] = base[(k0 + j) * rs + n];
  return u.v;
}

__global__ __launch_bounds__(256) void comirec_sa_kernel(
    const int* __restrict__ user_id, const int* __restrict__ history,
    const int* __restrict__ pos_item, const int* __restrict__ neg_items,
    const float* __restrict__ user_table, const float* __restrict__ item_table,
    const float* __restrict__ W1, const float* __restrict__ W2,
    const float* __restrict__ CW, float* __restrict__ out) {
  constexpr int L = 200, LP = 224, D = 64, HID = 256, NNEG = 100, NI = 101;
  constexpr int HR = 72;    // histA row stride (halves): 64 + 8 pad (bank spread)
  constexpr int W1R = 72;   // W1T row stride
  constexpr int W2R = 264;  // W2T row stride
  constexpr int TBR = 40;   // tanh staging row stride
  constexpr int ATR = 224;  // logits row stride (f32)
  constexpr int A16R = 232; // softmaxed-A f16 row stride

  __shared__ _Float16 histA[LP * HR];     // hist rows (A of GEMM1, B of GEMM3)
  __shared__ _Float16 W1T[HID * W1R];     // W1 columns contiguous
  __shared__ _Float16 W2T[16 * W2R];      // W2 columns contiguous, padded to N=16
  __shared__ _Float16 tbuf[8][16 * TBR];  // per-wave tanh(H) 16x32 staging
  __shared__ _Float16 A16[16 * A16R];     // softmaxed attention, transposed [k][l]
  __shared__ float AattT[4 * ATR];        // raw logits transposed [k][l]
  __shared__ float inter[4 * D];
  __shared__ float ue[4 * D];
  __shared__ float userv[D], posv[D];
  __shared__ int histIds[LP];
  __shared__ float dots[4];
  __shared__ int kbest;
  __shared__ float bnorm;

  const int b = blockIdx.x;
  const int tid = threadIdx.x;
  const int w = tid >> 5;

  // ---- phase 0: ids + per-b vectors ----
  if (tid < LP) histIds[tid] = (tid < L) ? history[(size_t)b * L + tid] : 0;
  if (tid >= 128 && tid < 192)
    userv[tid - 128] = user_table[(size_t)user_id[b] * D + (tid - 128)];
  if (tid >= 192)
    posv[tid - 192] = item_table[(size_t)pos_item[b] * D + (tid - 192)];
  __syncthreads();

  // ---- phase 1: stage f32 -> f16 tiles into LDS (float2 loads, packed stores) ----
  for (int i = tid; i < L * (D / 2); i += 256) {          // history rows
    int r = i >> 5, c2 = (i & 31) * 2;
    float2 v = *(const float2*)(item_table + (size_t)histIds[r] * D + c2);
    union { _Float16 h[2]; unsigned u; } p;
    p.h[0] = (_Float16)v.x;
    p.h[1] = (_Float16)v.y;
    *(unsigned*)&histA[r * HR + c2] = p.u;                // 4B aligned (HR even)
  }
  for (int i = tid; i < LP * 8; i += 256)                 // K-pad cols 64..71
    histA[(i >> 3) * HR + 64 + (i & 7)] = (_Float16)0.f;
  for (int i = tid; i < (LP - L) * D; i += 256)           // pad rows 200..223
    histA[(L + (i >> 6)) * HR + (i & 63)] = (_Float16)0.f;

  for (int i = tid; i < D * (HID / 2); i += 256) {        // W1 -> W1T (transpose)
    int k = i >> 7, h2 = (i & 127) * 2;
    float2 v = *(const float2*)(W1 + k * HID + h2);
    W1T[h2 * W1R + k] = (_Float16)v.x;
    W1T[(h2 + 1) * W1R + k] = (_Float16)v.y;
  }
  for (int i = tid; i < HID * 8; i += 256)                // zero K-pad of W1T
    W1T[(i >> 3) * W1R + 64 + (i & 7)] = (_Float16)0.f;
  for (int i = tid; i < 16 * W2R; i += 256) {
    int n = i / W2R, hh = i - n * W2R;
    W2T[i] = (_Float16)((n < 4 && hh < HID) ? W2[hh * 4 + n] : 0.f);
  }
  for (int i = tid; i < 16 * A16R; i += 256) A16[i] = (_Float16)0.f;
  __syncthreads();

  // ---- phase 2: fused  H = tanh(hist@W1),  logits = H@W2  (per-wave M tiles) ----
  for (int mt = w; mt < LP / 16; mt += 8) {
    const int r0 = ((tid >> 4) & 1) * 8, n = tid & 15;
    v16h a0 = load_a(histA, mt * 16, 0, HR);   // A frags reused across all hp
    v16h a1 = load_a(histA, mt * 16, 32, HR);
    v8f accA = {};
    for (int hp = 0; hp < 8; ++hp) {           // 32-wide HID chunks
      // 4 independent WMMAs back-to-back (two 16-wide N subtiles, K=64)
      v16h b00 = load_b(W1T, (hp * 2) * 16, 0, W1R);
      v16h b01 = load_b(W1T, (hp * 2) * 16, 32, W1R);
      v16h b10 = load_b(W1T, (hp * 2 + 1) * 16, 0, W1R);
      v16h b11 = load_b(W1T, (hp * 2 + 1) * 16, 32, W1R);
      v8f z = {};
      v8f h0 = wmma16(a0, b00, z);
      v8f h1 = wmma16(a0, b10, z);
      h0 = wmma16(a1, b01, h0);
      h1 = wmma16(a1, b11, h1);
      // branchless tanh, D-layout -> A-layout staging through per-wave LDS
#pragma unroll
      for (int r = 0; r < 8; ++r) {
        tbuf[w][(r0 + r) * TBR + n] = (_Float16)fast_tanh(h0[r]);
        tbuf[w][(r0 + r) * TBR + 16 + n] = (_Float16)fast_tanh(h1[r]);
      }
      v16h a2 = load_a(tbuf[w], 0, 0, TBR);    // tanh(H) chunk as A
      v16h b2 = load_b(W2T, 0, hp * 32, W2R);  // W2 chunk as B
      accA = wmma16(a2, b2, accA);
    }
    if (n < 4) {                               // only K=4 interest columns
#pragma unroll
      for (int r = 0; r < 8; ++r) AattT[n * ATR + mt * 16 + r0 + r] = accA[r];
    }
  }
  __syncthreads();

  // ---- phase 3: masked softmax over L (wave k handles interest k) ----
  if (w < 4) {
    int k = w, l = tid & 31;
    float mx = -3.0e38f;
    for (int i = l; i < L; i += 32) {
      float v = AattT[k * ATR + i];
      if (histIds[i] <= 0) v -= 1e9f;          // A + (-1e9)*(1-mask)
      AattT[k * ATR + i] = v;
      mx = fmaxf(mx, v);
    }
    for (int off = 16; off; off >>= 1) mx = fmaxf(mx, __shfl_xor(mx, off, 32));
    float s = 0.f;
    for (int i = l; i < L; i += 32) {
      float e = __expf(AattT[k * ATR + i] - mx);
      AattT[k * ATR + i] = e;
      s += e;
    }
    for (int off = 16; off; off >>= 1) s += __shfl_xor(s, off, 32);
    float inv = 1.0f / s;
    for (int i = l; i < L; i += 32)
      A16[k * A16R + i] = (_Float16)(AattT[k * ATR + i] * inv);
  }
  __syncthreads();

  // ---- phase 4: interests = A^T @ hist  (waves 0..3, one N tile each) ----
  if (w < 4) {
    int nt = w;
    v8f acc = {};
    for (int ks = 0; ks < 7; ++ks) {           // K = 224 (zero-padded past 200)
      v16h a = load_a(A16, 0, ks * 32, A16R);
      v16h bb = load_b_rowmajor(histA, nt * 16, ks * 32, HR);
      acc = wmma16(a, bb, acc);
    }
    int n = tid & 15;
    if ((tid & 31) < 16) {                     // D rows 0..3 = the K interests
#pragma unroll
      for (int r = 0; r < 4; ++r) inter[r * D + nt * 16 + n] = acc[r];
    }
  }
  __syncthreads();

  // ---- phase 5: user_embedding = [user_emb | interests] @ convert_W  (fp32) ----
  {
    int k = tid >> 6, d = tid & 63;
    float s = 0.f;
    for (int e = 0; e < D; ++e) s += userv[e] * CW[e * D + d];
    for (int e = 0; e < D; ++e) s += inter[k * D + e] * CW[(D + e) * D + d];
    ue[k * D + d] = s;
  }
  __syncthreads();

  // ---- phase 6: pick best interest by dot with positive item ----
  if (tid < 4) {
    float s = 0.f;
    for (int d2 = 0; d2 < D; ++d2) s += ue[tid * D + d2] * posv[d2];
    dots[tid] = s;
  }
  __syncthreads();
  if (tid == 0) {
    int kb = 0;
    float bv = dots[0];
    for (int k2 = 1; k2 < 4; ++k2)
      if (dots[k2] > bv) { bv = dots[k2]; kb = k2; }  // first-max like argmax
    kbest = kb;
    float bn = 0.f;
    for (int d2 = 0; d2 < D; ++d2) { float x = ue[kb * D + d2]; bn += x * x; }
    bnorm = sqrtf(bn);
  }
  __syncthreads();

  // ---- phase 7: cosine similarity vs pos + 100 neg items ----
  if (tid < NI) {
    int id = (tid == 0) ? pos_item[b] : neg_items[(size_t)b * NNEG + (tid - 1)];
    const float* iv = item_table + (size_t)id * D;
    const float* bvp = ue + kbest * D;
    float num = 0.f, nn = 0.f;
    for (int d2 = 0; d2 < D; ++d2) {
      float x = iv[d2];
      num += bvp[d2] * x;
      nn += x * x;
    }
    out[(size_t)b * NI + tid] =
        num / (fmaxf(bnorm, 1e-8f) * fmaxf(sqrtf(nn), 1e-8f));
  }
}

extern "C" void kernel_launch(void* const* d_in, const int* in_sizes, int n_in,
                              void* d_out, int out_size, void* d_ws, size_t ws_size,
                              hipStream_t stream) {
  (void)n_in; (void)out_size; (void)d_ws; (void)ws_size;
  const int* user_id    = (const int*)d_in[0];
  const int* history    = (const int*)d_in[1];
  const int* pos_item   = (const int*)d_in[2];
  const int* neg_items  = (const int*)d_in[3];
  const float* user_tab = (const float*)d_in[4];
  const float* item_tab = (const float*)d_in[5];
  const float* W1       = (const float*)d_in[6];
  const float* W2       = (const float*)d_in[7];
  const float* CW       = (const float*)d_in[8];
  float* out            = (float*)d_out;
  const int B = in_sizes[0];

  comirec_sa_kernel<<<dim3(B), dim3(256), 0, stream>>>(
      user_id, history, pos_item, neg_items, user_tab, item_tab, W1, W2, CW, out);
}
